// Decoders_26250840113393
// MI455X (gfx1250) — compile-verified, matrix operand order checked by reading, not run on hardware
//
#include <hip/hip_runtime.h>

#define N_PTS (1 << 20)
#define NPB   128            // points (and threads) per block

typedef _Float16 h8   __attribute__((ext_vector_type(8)));
typedef _Float16 v16h __attribute__((ext_vector_type(16)));
typedef float    v8f  __attribute__((ext_vector_type(8)));

// ---- pre-swizzled f16 weight-fragment layout in d_ws (element offsets) ----
#define SZ_W0   (3 * 64 * 32)   // K=96 (68 pad), N=64
#define SZ_W1   (2 * 64 * 32)   // K=64, N=64
#define SZ_WO   (2 * 48 * 32)   // K=64, N=48 (33 pad)
#define SZ_WC0  (3 * 64 * 32)   // K=96 (68 pad), N=64
#define SZ_WC1  (2 * 32 * 32)   // K=64, N=32
#define SZ_WCO  (1 * 16 * 32)   // K=32, N=16 (3 pad)
#define OFF_W0   0
#define OFF_W1   (OFF_W0 + SZ_W0)
#define OFF_WO   (OFF_W1 + SZ_W1)
#define OFF_WC0  (OFF_WO + SZ_WO)
#define OFF_WC1  (OFF_WC0 + SZ_WC0)
#define OFF_WCO  (OFF_WC1 + SZ_WC1)
#define WS_TOTAL (OFF_WCO + SZ_WCO)   // 22016 halves = 44 KB

__constant__ float c_hres[16] = {16.f, 19.f, 24.f, 30.f, 37.f, 46.f, 57.f, 71.f,
                                 89.f, 110.f, 136.f, 169.f, 210.f, 260.f, 322.f, 400.f};

struct Params {
  const float* p;
  const float* plane[12];
  const float* hash;
  const float* b0;  const float* b1;  const float* bo;
  const float* bc0; const float* bc1; const float* bco;
  const _Float16* wf;
  float* out;
};

// ---------------------------------------------------------------------------
// Weight prep: f32 -> f16, K padded to 32, per-(chunk,col) contiguous 32-half
// blocks so a lane's fragment is two aligned b128 loads. W0 rows permuted so
// the MLP input layout is [hash(0..31) | plane(32..67) | zero pad].
// ---------------------------------------------------------------------------
__global__ __launch_bounds__(256) void prep_weights_k(
    const float* W0, const float* W1, const float* Wo,
    const float* Wc0, const float* Wc1, const float* Wco, _Float16* wf) {
  int i = blockIdx.x * 256 + threadIdx.x;
  if (i >= WS_TOTAL) return;
  int base, N, srcN; const float* src;
  if (i < OFF_W1)       { base = OFF_W0;  N = 64; src = W0;  srcN = 64; }
  else if (i < OFF_WO)  { base = OFF_W1;  N = 64; src = W1;  srcN = 64; }
  else if (i < OFF_WC0) { base = OFF_WO;  N = 48; src = Wo;  srcN = 33; }
  else if (i < OFF_WC1) { base = OFF_WC0; N = 64; src = Wc0; srcN = 64; }
  else if (i < OFF_WCO) { base = OFF_WC1; N = 32; src = Wc1; srcN = 32; }
  else                  { base = OFF_WCO; N = 16; src = Wco; srcN = 3;  }
  int local = i - base;
  int kc = local / (N * 32);
  int n  = (local / 32) % N;
  int k  = local % 32;
  int kk = kc * 32 + k;
  int row = kk;
  bool valid = (n < srcN);
  if (base == OFF_W0) {                    // reorder input rows: hash first
    if (kk < 32)       row = 36 + kk;      // hash-feature weight rows
    else if (kk < 68)  row = kk - 32;      // plane-feature weight rows
    else               valid = false;      // K padding
  } else if (base == OFF_WC0) {
    if (kk >= 68) valid = false;           // K padding (geo+plane = 68)
  }
  float v = valid ? src[(size_t)row * srcN + n] : 0.f;
  wf[i] = (_Float16)v;
}

// ---------------------------------------------------------------------------
// Fragment load: 32 consecutive K-halves; lane's half picks kbase in {0,8}.
// A/B 16-bit fragment: elems 0..7 = K kb..kb+7, elems 8..15 = K kb+16..kb+23.
// ---------------------------------------------------------------------------
__device__ __forceinline__ v16h load_frag(const _Float16* base, int kb) {
  h8 lo = *(const h8*)(base + kb);
  h8 hi = *(const h8*)(base + 16 + kb);
  return __builtin_shufflevector(lo, hi, 0, 1, 2, 3, 4, 5, 6, 7,
                                 8, 9, 10, 11, 12, 13, 14, 15);
}

template <int KC, int NT>
__device__ __forceinline__ void mlp_layer(const v16h* A, const _Float16* wf,
                                          int Ncols, const float* bias, int biasN,
                                          v8f* acc, int kb, int nIdx) {
#pragma unroll
  for (int nt = 0; nt < NT; ++nt) {
    int col = nt * 16 + nIdx;
    float b = (col < biasN) ? bias[col] : 0.f;
    v8f c;
#pragma unroll
    for (int r = 0; r < 8; ++r) c[r] = b;
#pragma unroll
    for (int kc = 0; kc < KC; ++kc) {
      v16h B = load_frag(wf + (size_t)(kc * Ncols + col) * 32, kb);
      c = __builtin_amdgcn_wmma_f32_16x16x32_f16(false, A[kc], false, B,
                                                 (short)0, c, false, false);
    }
    acc[nt] = c;
  }
}

template <int NT>
__device__ __forceinline__ void store_act(_Float16* dst, int W, const v8f* acc,
                                          int hf, int nIdx, bool relu) {
#pragma unroll
  for (int nt = 0; nt < NT; ++nt) {
#pragma unroll
    for (int r = 0; r < 8; ++r) {
      float v = acc[nt][r];
      if (relu) v = fmaxf(v, 0.f);
      dst[(size_t)(r + 8 * hf) * W + nt * 16 + nIdx] = (_Float16)v;
    }
  }
}

// Bilinear sample, align_corners=True, border clamp; plane is (3, R, R).
__device__ __forceinline__ void bilerp3(const float* pl, int R, float u, float v,
                                        float* o) {
  float lim = (float)(R - 1);
  float x = fminf(fmaxf((u + 1.f) * 0.5f * lim, 0.f), lim);
  float y = fminf(fmaxf((v + 1.f) * 0.5f * lim, 0.f), lim);
  float xf = floorf(x), yf = floorf(y);
  float wx = x - xf, wy = y - yf;
  int x0 = (int)xf, y0 = (int)yf;
  int x1 = min(x0 + 1, R - 1), y1 = min(y0 + 1, R - 1);
  float w00 = (1.f - wy) * (1.f - wx), w01 = (1.f - wy) * wx;
  float w10 = wy * (1.f - wx),         w11 = wy * wx;
  int s = R * R;
#pragma unroll
  for (int ch = 0; ch < 3; ++ch) {
    const float* Pc = pl + ch * s;
    o[ch] = Pc[y0 * R + x0] * w00 + Pc[y0 * R + x1] * w01 +
            Pc[y1 * R + x0] * w10 + Pc[y1 * R + x1] * w11;
  }
}

__global__ __launch_bounds__(NPB) void fused_field_kernel(Params P) {
  // Activation rows in A-fragment-friendly layout: [hash 0..31 | plane 32..67 | 0]
  __shared__ _Float16 sFeat[NPB][96];          // 24.0 KB
  __shared__ _Float16 sB[NPB / 32][16 * 64];   //  8.0 KB inter-layer bounce

  const int tid = threadIdx.x;

  // ---------------- Phase 1: per-point feature gathers -------------------
  {
    const size_t g = (size_t)blockIdx.x * NPB + tid;
    float x = P.p[g * 3 + 0], y = P.p[g * 3 + 1], z = P.p[g * 3 + 2];
    float nx = x * 0.25f, ny = y * 0.25f, nz = z * 0.5f;   // -> [-1,1]
    const float uu[3] = {nx, nx, ny};
    const float vv[3] = {ny, nz, nz};
    const int RES[4] = {128, 256, 512, 1024};
#pragma unroll
    for (int li = 0; li < 4; ++li) {
#pragma unroll
      for (int pi = 0; pi < 3; ++pi) {
        float o[3];
        bilerp3(P.plane[li * 3 + pi], RES[li], uu[pi], vv[pi], o);
#pragma unroll
        for (int ch = 0; ch < 3; ++ch)
          sFeat[tid][32 + li * 9 + pi * 3 + ch] = (_Float16)o[ch];
      }
    }
    float px = (nx + 1.f) * 0.5f, py = (ny + 1.f) * 0.5f, pz = (nz + 1.f) * 0.5f;
    for (int l = 0; l < 16; ++l) {
      float r = c_hres[l];
      float fx = px * r, fy = py * r, fz = pz * r;
      float x0f = floorf(fx), y0f = floorf(fy), z0f = floorf(fz);
      float wx = fx - x0f, wy = fy - y0f, wz = fz - z0f;
      unsigned xi = (unsigned)x0f, yi = (unsigned)y0f, zi = (unsigned)z0f;
      float f0 = 0.f, f1 = 0.f;
#pragma unroll
      for (int dz = 0; dz < 2; ++dz)
#pragma unroll
        for (int dy = 0; dy < 2; ++dy)
#pragma unroll
          for (int dx = 0; dx < 2; ++dx) {
            unsigned h = (xi + (unsigned)dx) ^
                         ((yi + (unsigned)dy) * 2654435761u) ^
                         ((zi + (unsigned)dz) * 805459861u);
            const float* tt = P.hash + ((size_t)l * 65536u + (h & 65535u)) * 2;
            float w = (dx ? wx : 1.f - wx) * (dy ? wy : 1.f - wy) *
                      (dz ? wz : 1.f - wz);
            f0 += tt[0] * w;
            f1 += tt[1] * w;
          }
      sFeat[tid][l * 2 + 0] = (_Float16)f0;
      sFeat[tid][l * 2 + 1] = (_Float16)f1;
    }
#pragma unroll
    for (int j = 68; j < 96; ++j) sFeat[tid][j] = (_Float16)0.f;
  }
  __syncthreads();

  // ---------------- Phase 2: WMMA MLPs, 16-point tiles per wave ----------
  const int wave = tid >> 5;
  const int lane = tid & 31;
  const int hf   = lane >> 4;     // which K-half this lane carries
  const int nIdx = lane & 15;     // B/C/D column, A row
  const int kb   = hf * 8;
  _Float16* sBw = &sB[wave][0];

  for (int t = 0; t < 2; ++t) {
    const int pbase = (wave * 2 + t) * 16;
    const size_t gbase = (size_t)blockIdx.x * NPB + pbase;
    const _Float16* featRow = &sFeat[pbase + nIdx][0];
    const _Float16* bRow64 = sBw + nIdx * 64;
    v16h A[3];
    v8f acc[4];

    // --- sigma MLP: 68(->96) -> 64 -> 64 -> 33(->48) tanh ---
#pragma unroll
    for (int c = 0; c < 3; ++c) A[c] = load_frag(featRow + c * 32, kb);
    mlp_layer<3, 4>(A, P.wf + OFF_W0, 64, P.b0, 64, acc, kb, nIdx);
    store_act<4>(sBw, 64, acc, hf, nIdx, true);
#pragma unroll
    for (int c = 0; c < 2; ++c) A[c] = load_frag(bRow64 + c * 32, kb);
    mlp_layer<2, 4>(A, P.wf + OFF_W1, 64, P.b1, 64, acc, kb, nIdx);
    store_act<4>(sBw, 64, acc, hf, nIdx, true);
#pragma unroll
    for (int c = 0; c < 2; ++c) A[c] = load_frag(bRow64 + c * 32, kb);
    mlp_layer<2, 3>(A, P.wf + OFF_WO, 48, P.bo, 33, acc, kb, nIdx);
#pragma unroll
    for (int nt = 0; nt < 3; ++nt)
#pragma unroll
      for (int r = 0; r < 8; ++r) acc[nt][r] = tanhf(acc[nt][r]);
    if (nIdx == 0) {                       // sdf = column 0 -> out[...,3]
#pragma unroll
      for (int r = 0; r < 8; ++r)
        P.out[(gbase + r + 8 * hf) * 4 + 3] = acc[0][r];
    }
#pragma unroll
    for (int nt = 0; nt < 3; ++nt) {       // geo = cols 1..32, overwrite hash slots
      int col = nt * 16 + nIdx;
      if (col >= 1 && col < 33) {
#pragma unroll
        for (int r = 0; r < 8; ++r)
          sFeat[pbase + r + 8 * hf][col - 1] = (_Float16)acc[nt][r];
      }
    }

    // --- color MLP: [geo|plane]=68(->96) -> 64 -> 32 -> 3(->16) sigmoid ---
#pragma unroll
    for (int c = 0; c < 3; ++c) A[c] = load_frag(featRow + c * 32, kb);
    mlp_layer<3, 4>(A, P.wf + OFF_WC0, 64, P.bc0, 64, acc, kb, nIdx);
    store_act<4>(sBw, 64, acc, hf, nIdx, true);
#pragma unroll
    for (int c = 0; c < 2; ++c) A[c] = load_frag(bRow64 + c * 32, kb);
    mlp_layer<2, 2>(A, P.wf + OFF_WC1, 32, P.bc1, 32, acc, kb, nIdx);
    store_act<2>(sBw, 32, acc, hf, nIdx, true);
    A[0] = load_frag(sBw + nIdx * 32, kb);
    mlp_layer<1, 1>(A, P.wf + OFF_WCO, 16, P.bco, 3, acc, kb, nIdx);
    if (nIdx < 3) {
#pragma unroll
      for (int r = 0; r < 8; ++r) {
        float v = acc[0][r];
        v = 1.f / (1.f + expf(-v));
        P.out[(gbase + r + 8 * hf) * 4 + nIdx] = v;
      }
    }
  }
}

extern "C" void kernel_launch(void* const* d_in, const int* in_sizes, int n_in,
                              void* d_out, int out_size, void* d_ws, size_t ws_size,
                              hipStream_t stream) {
  (void)in_sizes; (void)n_in; (void)out_size; (void)ws_size;
  _Float16* wf = (_Float16*)d_ws;   // 44 KB of fragment-layout f16 weights

  prep_weights_k<<<(WS_TOTAL + 255) / 256, 256, 0, stream>>>(
      (const float*)d_in[14], (const float*)d_in[16], (const float*)d_in[18],
      (const float*)d_in[20], (const float*)d_in[22], (const float*)d_in[24], wf);

  Params P;
  P.p = (const float*)d_in[0];
  for (int i = 0; i < 12; ++i) P.plane[i] = (const float*)d_in[1 + i];
  P.hash = (const float*)d_in[13];
  P.b0  = (const float*)d_in[15];
  P.b1  = (const float*)d_in[17];
  P.bo  = (const float*)d_in[19];
  P.bc0 = (const float*)d_in[21];
  P.bc1 = (const float*)d_in[23];
  P.bco = (const float*)d_in[25];
  P.wf = wf;
  P.out = (float*)d_out;

  fused_field_kernel<<<N_PTS / NPB, NPB, 0, stream>>>(P);
}
